// Transformer_68796786147596
// MI455X (gfx1250) — matrix operand order repaired and checked by reading, compiled, not run on hardware
//
#include <hip/hip_runtime.h>
#include <math.h>
#include <stdint.h>

// ---------------- problem constants (match reference) ----------------
#define BB    2
#define SS    1024            // SE == SD == 1024
#define DD    512             // model dim
#define HH    8               // heads
#define DHH   64              // head dim
#define MMID  512             // H*DH
#define FFH   2048            // SwiGLU hidden
#define VV    32000           // vocab
#define NLE   2
#define NLD   2
#define BSR   (BB*SS)         // 2048 token rows
#define NBH   (BB*HH)         // 16 attention batches
#define EPSV  1e-5f

typedef __attribute__((ext_vector_type(16))) _Float16 v16h;
typedef __attribute__((ext_vector_type(8)))  _Float16 v8h;
typedef __attribute__((ext_vector_type(8)))  float    v8f;
typedef int v4i __attribute__((__vector_size__(16)));
typedef _Float16 f16;

#define LDS_PAD 8   // f16 row padding: keeps 16B alignment, staggers banks

// gfx1250 async global->LDS copy path (ASYNCcnt-tracked); guarded so the
// file still compiles (sync fallback) if the toolchain lacks the builtins.
// Probe-confirmed param types: (int4 AS1* src, int4 AS3* dst, imm, imm cpol).
#if defined(__AMDGCN__) && \
    __has_builtin(__builtin_amdgcn_global_load_async_to_lds_b128) && \
    __has_builtin(__builtin_amdgcn_s_wait_asynccnt)
#define USE_ASYNC 1
#define ASYNC_CP16(gp, lp)                                            \
    __builtin_amdgcn_global_load_async_to_lds_b128(                   \
        (__attribute__((address_space(1))) v4i*)(gp),                 \
        (__attribute__((address_space(3))) v4i*)(lp), 0, 0)
#else
#define USE_ASYNC 0
#endif

// =====================================================================
// Batched LDS-staged WMMA GEMM: C[m,n] = scale * sum_k A[m,k]*B[k,n] + bias[n]
//   A element (m,k) at A[m*lda + k]                       (f16)
//   B element (k,n) at B[n*ldb + k]   (ALWAYS k-contiguous per column:
//       weights are stored pre-transposed, V is written transposed)
//   C f32 at C[m*ldc + n];  Ch f16 at Ch[m*ldch+n] or (chTrans) Ch[n*ldch+m]
// Batch z: bi = z/batchH, hi = z%batchH; per-operand (bi,hi) stride pairs.
// Block: 256 threads = 8 waves; block tile 256(M) x 64(N); wave tile 32x64.
// Double-buffered LDS k-tiles (32 deep); stage-in is contiguous 16B copies
// (async global->LDS when available). 8 WMMAs per wave per K-step.
// =====================================================================
__global__ __launch_bounds__(256) void wmma_gemm_kernel(
    const f16* __restrict__ A, const f16* __restrict__ B,
    float* __restrict__ C, f16* __restrict__ Ch,
    const float* __restrict__ bias,
    int Kdim, int lda, int ldb, int ldc, int ldch, int chTrans, float scale,
    int batchH,
    long long sAb, long long sAh, long long sBb, long long sBh,
    long long sCb, long long sCh)
{
    __shared__ f16 lA[2][256][32 + LDS_PAD];   // 40 KB
    __shared__ f16 lB[2][64][32 + LDS_PAD];    // 10 KB

    const int z  = blockIdx.z;
    const int bi = z / batchH;
    const int hi = z - bi * batchH;
    A += (long long)bi * sAb + (long long)hi * sAh;
    B += (long long)bi * sBb + (long long)hi * sBh;
    const long long cOff = (long long)bi * sCb + (long long)hi * sCh;

    const int tid  = threadIdx.x;
    const int lane = tid & 31;
    const int wave = tid >> 5;
    const int r    = lane & 15;      // row within A-frag / col within B-frag
    const int hg   = lane >> 4;      // lane half-group

    const int m0 = blockIdx.y * 256;
    const int n0 = blockIdx.x * 64;

    // cooperative staging coordinates: thread owns one 32-k A row (64B)
    // and one 16B B segment. All contiguous, 16B aligned.
    const f16* gA = A + (long long)(m0 + tid) * lda;
    const int  bn = tid >> 2;
    const int  bs = (tid & 3) * 8;
    const f16* gB = B + (long long)(n0 + bn) * ldb + bs;

    auto stage = [&](int k0, int buf) {
#if USE_ASYNC
        ASYNC_CP16(gA + k0,      &lA[buf][tid][0]);
        ASYNC_CP16(gA + k0 + 8,  &lA[buf][tid][8]);
        ASYNC_CP16(gA + k0 + 16, &lA[buf][tid][16]);
        ASYNC_CP16(gA + k0 + 24, &lA[buf][tid][24]);
        ASYNC_CP16(gB + k0,      &lB[buf][bn][bs]);
#else
        *(v8h*)&lA[buf][tid][0]  = *(const v8h*)(gA + k0);
        *(v8h*)&lA[buf][tid][8]  = *(const v8h*)(gA + k0 + 8);
        *(v8h*)&lA[buf][tid][16] = *(const v8h*)(gA + k0 + 16);
        *(v8h*)&lA[buf][tid][24] = *(const v8h*)(gA + k0 + 24);
        *(v8h*)&lB[buf][bn][bs]  = *(const v8h*)(gB + k0);
#endif
    };

    v8f acc0[4] = {};
    v8f acc1[4] = {};

    stage(0, 0);
    int cur = 0;
    for (int k0 = 0; k0 < Kdim; k0 += 32) {
#if USE_ASYNC
        __builtin_amdgcn_s_wait_asynccnt(0);
#endif
        __syncthreads();
        if (k0 + 32 < Kdim) stage(k0 + 32, cur ^ 1);

        // ---- A fragments (ISA 7.12.2): lane r = row M, two 16B chunks each
        const int row = wave * 32;
        const v8h a0lo = *(const v8h*)&lA[cur][row + r][hg * 8];
        const v8h a0hi = *(const v8h*)&lA[cur][row + r][16 + hg * 8];
        const v8h a1lo = *(const v8h*)&lA[cur][row + 16 + r][hg * 8];
        const v8h a1hi = *(const v8h*)&lA[cur][row + 16 + r][16 + hg * 8];
        const v16h a0 = __builtin_shufflevector(a0lo, a0hi,
                            0, 1, 2, 3, 4, 5, 6, 7, 8, 9, 10, 11, 12, 13, 14, 15);
        const v16h a1 = __builtin_shufflevector(a1lo, a1hi,
                            0, 1, 2, 3, 4, 5, 6, 7, 8, 9, 10, 11, 12, 13, 14, 15);

        // ---- B fragments: lane = column n, same K order as A: two 16B chunks
        v16h bfr[4];
        #pragma unroll
        for (int t = 0; t < 4; ++t) {
            const v8h blo = *(const v8h*)&lB[cur][t * 16 + r][hg * 8];
            const v8h bhi = *(const v8h*)&lB[cur][t * 16 + r][16 + hg * 8];
            bfr[t] = __builtin_shufflevector(blo, bhi,
                         0, 1, 2, 3, 4, 5, 6, 7, 8, 9, 10, 11, 12, 13, 14, 15);
        }
        #pragma unroll
        for (int t = 0; t < 4; ++t)
            acc0[t] = __builtin_amdgcn_wmma_f32_16x16x32_f16(
                          false, a0, false, bfr[t], (short)0, acc0[t], false, false);
        #pragma unroll
        for (int t = 0; t < 4; ++t)
            acc1[t] = __builtin_amdgcn_wmma_f32_16x16x32_f16(
                          false, a1, false, bfr[t], (short)0, acc1[t], false, false);

        cur ^= 1;
    }

    // ---- store: C/D f32 layout: m = 8*hg + v (+16 for acc1), n = 16*t + r
    #pragma unroll
    for (int t = 0; t < 4; ++t) {
        const int n = n0 + t * 16 + r;
        const float bval = bias ? bias[n] : 0.0f;
        #pragma unroll
        for (int v = 0; v < 8; ++v) {
            const int mlo = m0 + wave * 32 + hg * 8 + v;
            const int mhi = mlo + 16;
            const float v0 = acc0[t][v] * scale + bval;
            const float v1 = acc1[t][v] * scale + bval;
            if (C) {
                C[cOff + (long long)mlo * ldc + n] = v0;
                C[cOff + (long long)mhi * ldc + n] = v1;
            }
            if (Ch) {
                if (chTrans) {
                    Ch[cOff + (long long)n * ldch + mlo] = (f16)v0;
                    Ch[cOff + (long long)n * ldch + mhi] = (f16)v1;
                } else {
                    Ch[cOff + (long long)mlo * ldch + n] = (f16)v0;
                    Ch[cOff + (long long)mhi * ldch + n] = (f16)v1;
                }
            }
        }
    }
}

// ---------------- embedding + sinusoidal positional encoding ----------------
__global__ void embed_pe_kernel(const int* __restrict__ tok,
                                const float* __restrict__ emb,
                                float* __restrict__ x, f16* __restrict__ xh)
{
    const size_t i = (size_t)blockIdx.x * blockDim.x + threadIdx.x;
    if (i >= (size_t)BSR * DD) return;
    const int    d  = (int)(i % DD);
    const size_t bs = i / DD;
    const int    s  = (int)(bs % SS);
    const int    tk = tok[bs];
    const float freq = __expf(-(float)(d & ~1) * (9.210340372f / (float)DD));
    const float ang  = (float)s * freq;
    const float pe   = (d & 1) ? __cosf(ang) : __sinf(ang);
    const float val  = emb[(size_t)tk * DD + d] + pe;
    x[i]  = val;
    xh[i] = (f16)val;
}

// ---------------- RMSNorm(in1 + in2) * g; writes f32 + f16 -------------------
__global__ __launch_bounds__(256) void rmsnorm_kernel(
    const float* __restrict__ in1, const float* __restrict__ in2,
    const float* __restrict__ g, float* __restrict__ out, f16* __restrict__ outh)
{
    __shared__ float red[256];
    const int row = blockIdx.x;
    const int t   = threadIdx.x;
    const float* p1 = in1 + (size_t)row * DD;
    const float* p2 = in2 + (size_t)row * DD;
    const float v0 = p1[t]       + p2[t];
    const float v1 = p1[t + 256] + p2[t + 256];
    red[t] = v0 * v0 + v1 * v1;
    __syncthreads();
    for (int s = 128; s > 0; s >>= 1) {
        if (t < s) red[t] += red[t + s];
        __syncthreads();
    }
    const float rms = rsqrtf(red[0] / (float)DD + EPSV);
    const float o0 = g[t]       * v0 * rms;
    const float o1 = g[t + 256] * v1 * rms;
    out [(size_t)row * DD + t]        = o0;
    out [(size_t)row * DD + t + 256]  = o1;
    outh[(size_t)row * DD + t]        = (f16)o0;
    outh[(size_t)row * DD + t + 256]  = (f16)o1;
}

// ---------------- row softmax with optional causal mask ----------------------
__global__ __launch_bounds__(256) void softmax_kernel(
    const float* __restrict__ scores, f16* __restrict__ probs, int causal)
{
    __shared__ float red[256];
    const int row = blockIdx.x;
    const int z   = blockIdx.y;
    const float* sp = scores + ((size_t)z * SS + row) * SS;
    f16*         pp = probs  + ((size_t)z * SS + row) * SS;
    const int t = threadIdx.x;

    float v[4];
    float mx = -1e30f;
    #pragma unroll
    for (int i = 0; i < 4; ++i) {
        const int c = t + i * 256;
        float x = sp[c];
        if (causal && c > row) x = -1e30f;
        v[i] = x;
        mx = fmaxf(mx, x);
    }
    red[t] = mx; __syncthreads();
    for (int s = 128; s > 0; s >>= 1) {
        if (t < s) red[t] = fmaxf(red[t], red[t + s]);
        __syncthreads();
    }
    mx = red[0];
    __syncthreads();
    float sum = 0.f;
    #pragma unroll
    for (int i = 0; i < 4; ++i) { v[i] = __expf(v[i] - mx); sum += v[i]; }
    red[t] = sum; __syncthreads();
    for (int s = 128; s > 0; s >>= 1) {
        if (t < s) red[t] += red[t + s];
        __syncthreads();
    }
    const float inv = 1.0f / red[0];
    #pragma unroll
    for (int i = 0; i < 4; ++i) pp[t + i * 256] = (f16)(v[i] * inv);
}

// ---------------- SwiGLU gate: hm = silu(h1) * h3 (f16 out) ------------------
__global__ void silu_mul_kernel(const float* __restrict__ h1,
                                const float* __restrict__ h3,
                                f16* __restrict__ hm, size_t n)
{
    const size_t i = (size_t)blockIdx.x * blockDim.x + threadIdx.x;
    if (i >= n) return;
    const float s   = h1[i];
    const float sig = 1.0f / (1.0f + __expf(-s));
    hm[i] = (f16)(s * sig * h3[i]);
}

// ------- weight prep: fp32 [L][K][N] -> fp16 transposed [L][N][K] ------------
__global__ void transpose_f32_to_f16_kernel(const float* __restrict__ in,
                                            f16* __restrict__ out,
                                            int K, int N, size_t total)
{
    const size_t i = (size_t)blockIdx.x * blockDim.x + threadIdx.x;
    if (i >= total) return;
    const size_t kn  = (size_t)K * N;
    const size_t l   = i / kn;
    const size_t rem = i - l * kn;
    const int n = (int)(rem / K);
    const int k = (int)(rem - (size_t)n * K);
    out[i] = (f16)in[l * kn + (size_t)k * N + n];
}

// =====================================================================
// host-side helpers
// =====================================================================
static inline void launch_gemm(hipStream_t st,
    const f16* A, const f16* B, float* C, f16* Ch, const float* bias,
    int M, int N, int K, int lda, int ldb, int ldc, int ldch, int chTrans,
    float scale, int nz, int batchH,
    long long sAb, long long sAh, long long sBb, long long sBh,
    long long sCb, long long sCh)
{
    dim3 grid((unsigned)(N / 64), (unsigned)(M / 256), (unsigned)nz);
    wmma_gemm_kernel<<<grid, dim3(256), 0, st>>>(
        A, B, C, Ch, bias, K, lda, ldb, ldc, ldch, chTrans, scale,
        batchH, sAb, sAh, sBb, sBh, sCb, sCh);
}

static void run_attention(hipStream_t st, const f16* xq, const f16* xkv,
    const f16* WqT, const float* bq, const f16* WkT, const float* bk,
    const f16* WvT, const float* bv, const f16* WoT, const float* bo,
    f16* qb, f16* kb, f16* vb, f16* ob, float* scores, f16* probs,
    float* attn_out, int causal)
{
    // q/k projections -> row-major f16 [token][h*DH+d]
    launch_gemm(st, xq,  WqT, nullptr, qb, bq, BSR, MMID, DD, DD, DD,
                MMID, MMID, 0, 1.f, 1, 1, 0, 0, 0, 0, 0, 0);
    launch_gemm(st, xkv, WkT, nullptr, kb, bk, BSR, MMID, DD, DD, DD,
                MMID, MMID, 0, 1.f, 1, 1, 0, 0, 0, 0, 0, 0);
    // v projection -> TRANSPOSED f16 [h*DH+d][token] so P.V's B is k-contiguous
    launch_gemm(st, xkv, WvT, nullptr, vb, bv, BSR, MMID, DD, DD, DD,
                MMID, BSR, 1, 1.f, 1, 1, 0, 0, 0, 0, 0, 0);
    // scores[b,h] = q . k^T / sqrt(DH): batched over 16 (b,h)
    launch_gemm(st, qb, kb, scores, nullptr, nullptr, SS, SS, DHH,
                MMID, MMID, SS, SS, 0, 0.125f, NBH, HH,
                (long long)SS * MMID, DHH, (long long)SS * MMID, DHH,
                (long long)HH * SS * SS, (long long)SS * SS);
    softmax_kernel<<<dim3(SS, NBH), 256, 0, st>>>(scores, probs, causal);
    // o[b,h] = p . v : batched; B = vT: element (j,d) at vb[(h*DH+d)*BSR + b*SS + j]
    launch_gemm(st, probs, vb, nullptr, ob, nullptr, SS, DHH, SS,
                SS, BSR, MMID, MMID, 0, 1.f, NBH, HH,
                (long long)HH * SS * SS, (long long)SS * SS,
                (long long)SS, (long long)DHH * BSR,
                (long long)SS * MMID, DHH);
    // output projection: [BSR,MMID] x [MMID,DD] + bo -> f32
    launch_gemm(st, ob, WoT, attn_out, nullptr, bo, BSR, DD, MMID, MMID, MMID,
                DD, DD, 0, 1.f, 1, 1, 0, 0, 0, 0, 0, 0);
}

static void run_ffn(hipStream_t st, const f16* xin,
    const f16* w1T, const f16* w2T, const f16* w3T,
    float* h1, float* h3, f16* hm, float* ffn_out)
{
    launch_gemm(st, xin, w1T, h1, nullptr, nullptr, BSR, FFH, DD, DD, DD,
                FFH, FFH, 0, 1.f, 1, 1, 0, 0, 0, 0, 0, 0);
    launch_gemm(st, xin, w3T, h3, nullptr, nullptr, BSR, FFH, DD, DD, DD,
                FFH, FFH, 0, 1.f, 1, 1, 0, 0, 0, 0, 0, 0);
    const size_t n = (size_t)BSR * FFH;
    silu_mul_kernel<<<(unsigned)((n + 255) / 256), 256, 0, st>>>(h1, h3, hm, n);
    launch_gemm(st, hm, w2T, ffn_out, nullptr, nullptr, BSR, DD, FFH, FFH, FFH,
                DD, DD, 0, 1.f, 1, 1, 0, 0, 0, 0, 0, 0);
}

// =====================================================================
extern "C" void kernel_launch(void* const* d_in, const int* in_sizes, int n_in,
                              void* d_out, int out_size, void* d_ws, size_t ws_size,
                              hipStream_t stream)
{
    (void)in_sizes; (void)n_in; (void)out_size;

    // ---- inputs in setup_inputs() dict order (nested dicts flattened) ----
    const int*   in_seq  = (const int*)  d_in[0];
    const int*   out_seq = (const int*)  d_in[1];
    const float* emb_in  = (const float*)d_in[2];
    const float* emb_out = (const float*)d_in[3];
    const float* eWq=(const float*)d_in[4];  const float* ebq=(const float*)d_in[5];
    const float* eWk=(const float*)d_in[6];  const float* ebk=(const float*)d_in[7];
    const float* eWv=(const float*)d_in[8];  const float* ebv=(const float*)d_in[9];
    const float* eWo=(const float*)d_in[10]; const float* ebo=(const float*)d_in[11];
    const float* eg1=(const float*)d_in[12]; const float* eg2=(const float*)d_in[13];
    const float* ew1=(const float*)d_in[14]; const float* ew2=(const float*)d_in[15];
    const float* ew3=(const float*)d_in[16];
    const float* dWq1=(const float*)d_in[17]; const float* dbq1=(const float*)d_in[18];
    const float* dWk1=(const float*)d_in[19]; const float* dbk1=(const float*)d_in[20];
    const float* dWv1=(const float*)d_in[21]; const float* dbv1=(const float*)d_in[22];
    const float* dWo1=(const float*)d_in[23]; const float* dbo1=(const float*)d_in[24];
    const float* dWq2=(const float*)d_in[25]; const float* dbq2=(const float*)d_in[26];
    const float* dWk2=(const float*)d_in[27]; const float* dbk2=(const float*)d_in[28];
    const float* dWv2=(const float*)d_in[29]; const float* dbv2=(const float*)d_in[30];
    const float* dWo2=(const float*)d_in[31]; const float* dbo2=(const float*)d_in[32];
    const float* dg1=(const float*)d_in[33]; const float* dg2=(const float*)d_in[34];
    const float* dg3=(const float*)d_in[35];
    const float* dw1=(const float*)d_in[36]; const float* dw2=(const float*)d_in[37];
    const float* dw3=(const float*)d_in[38];
    const float* linW=(const float*)d_in[39]; const float* linb=(const float*)d_in[40];

    // ---- bump allocator over d_ws ----
    char*  base = (char*)d_ws;
    size_t off  = 0;
    auto get = [&](size_t bytes) -> void* {
        void* p = base + off;
        off = (off + bytes + 255) & ~(size_t)255;
        return p;
    };
    const size_t projW = (size_t)NLE * DD * MMID;   // same for NLD
    const size_t ffnW  = (size_t)NLE * DD * FFH;

    // f16 weight copies (stored TRANSPOSED: [L][N][K], k-contiguous)
    f16* eWq_h = (f16*)get(sizeof(f16) * projW);
    f16* eWk_h = (f16*)get(sizeof(f16) * projW);
    f16* eWv_h = (f16*)get(sizeof(f16) * projW);
    f16* eWo_h = (f16*)get(sizeof(f16) * projW);
    f16* ew1_h = (f16*)get(sizeof(f16) * ffnW);
    f16* ew2_h = (f16*)get(sizeof(f16) * ffnW);
    f16* ew3_h = (f16*)get(sizeof(f16) * ffnW);
    f16* dWq1_h = (f16*)get(sizeof(f16) * projW);
    f16* dWk1_h = (f16*)get(sizeof(f16) * projW);
    f16* dWv1_h = (f16*)get(sizeof(f16) * projW);
    f16* dWo1_h = (f16*)get(sizeof(f16) * projW);
    f16* dWq2_h = (f16*)get(sizeof(f16) * projW);
    f16* dWk2_h = (f16*)get(sizeof(f16) * projW);
    f16* dWv2_h = (f16*)get(sizeof(f16) * projW);
    f16* dWo2_h = (f16*)get(sizeof(f16) * projW);
    f16* dw1_h = (f16*)get(sizeof(f16) * ffnW);
    f16* dw2_h = (f16*)get(sizeof(f16) * ffnW);
    f16* dw3_h = (f16*)get(sizeof(f16) * ffnW);
    f16* linW_h = (f16*)get(sizeof(f16) * (size_t)DD * VV);

    // activations
    const size_t nTok = (size_t)BSR * DD;
    float* xe   = (float*)get(sizeof(float) * nTok);   // encoder stream / enc_out
    f16*   xeh  = (f16*)  get(sizeof(f16)   * nTok);
    float* xd   = (float*)get(sizeof(float) * nTok);   // decoder stream
    f16*   xdh  = (f16*)  get(sizeof(f16)   * nTok);
    float* tmpD = (float*)get(sizeof(float) * nTok);   // attn_out / ffn_out
    float* dn1  = (float*)get(sizeof(float) * nTok);   // post-norm intermediate
    f16*   dn1h = (f16*)  get(sizeof(f16)   * nTok);
    f16* qb = (f16*)get(sizeof(f16) * (size_t)BSR * MMID);
    f16* kb = (f16*)get(sizeof(f16) * (size_t)BSR * MMID);
    f16* vb = (f16*)get(sizeof(f16) * (size_t)BSR * MMID);  // [h*DH+d][token]
    f16* ob = (f16*)get(sizeof(f16) * (size_t)BSR * MMID);
    float* scores = (float*)get(sizeof(float) * (size_t)NBH * SS * SS);
    f16*   probs  = (f16*)  get(sizeof(f16)   * (size_t)NBH * SS * SS);
    float* h1 = (float*)get(sizeof(float) * (size_t)BSR * FFH);
    float* h3 = (float*)get(sizeof(float) * (size_t)BSR * FFH);
    f16*   hm = (f16*)  get(sizeof(f16)   * (size_t)BSR * FFH);

    if (off > ws_size) return;  // workspace too small: do nothing (defensive)

    // ---- weight prep: fp32 -> fp16 + transpose (once per call) ----
    auto tcvt = [&](const float* src, f16* dst, int L, int K, int N) {
        const size_t total = (size_t)L * K * N;
        transpose_f32_to_f16_kernel<<<(unsigned)((total + 255) / 256), 256, 0,
                                      stream>>>(src, dst, K, N, total);
    };
    tcvt(eWq, eWq_h, NLE, DD, MMID); tcvt(eWk, eWk_h, NLE, DD, MMID);
    tcvt(eWv, eWv_h, NLE, DD, MMID); tcvt(eWo, eWo_h, NLE, MMID, DD);
    tcvt(ew1, ew1_h, NLE, DD, FFH);  tcvt(ew2, ew2_h, NLE, FFH, DD);
    tcvt(ew3, ew3_h, NLE, DD, FFH);
    tcvt(dWq1, dWq1_h, NLD, DD, MMID); tcvt(dWk1, dWk1_h, NLD, DD, MMID);
    tcvt(dWv1, dWv1_h, NLD, DD, MMID); tcvt(dWo1, dWo1_h, NLD, MMID, DD);
    tcvt(dWq2, dWq2_h, NLD, DD, MMID); tcvt(dWk2, dWk2_h, NLD, DD, MMID);
    tcvt(dWv2, dWv2_h, NLD, DD, MMID); tcvt(dWo2, dWo2_h, NLD, MMID, DD);
    tcvt(dw1, dw1_h, NLD, DD, FFH); tcvt(dw2, dw2_h, NLD, FFH, DD);
    tcvt(dw3, dw3_h, NLD, DD, FFH);
    tcvt(linW, linW_h, 1, DD, VV);

    // ---- encoder ----
    embed_pe_kernel<<<(unsigned)((nTok + 255) / 256), 256, 0, stream>>>(
        in_seq, emb_in, xe, xeh);
    for (int i = 0; i < NLE; ++i) {
        const size_t po = (size_t)i * DD * MMID;
        const size_t fo = (size_t)i * DD * FFH;
        run_attention(stream, xeh, xeh,
                      eWq_h + po, ebq + i * MMID, eWk_h + po, ebk + i * MMID,
                      eWv_h + po, ebv + i * MMID, eWo_h + po, ebo + i * DD,
                      qb, kb, vb, ob, scores, probs, tmpD, /*causal=*/0);
        rmsnorm_kernel<<<BSR, 256, 0, stream>>>(xe, tmpD, eg1 + i * DD, dn1, dn1h);
        run_ffn(stream, dn1h, ew1_h + fo, ew2_h + fo, ew3_h + fo, h1, h3, hm, tmpD);
        rmsnorm_kernel<<<BSR, 256, 0, stream>>>(dn1, tmpD, eg2 + i * DD, xe, xeh);
    }
    // xe/xeh now hold enc_out

    // ---- decoder ----
    embed_pe_kernel<<<(unsigned)((nTok + 255) / 256), 256, 0, stream>>>(
        out_seq, emb_out, xd, xdh);
    for (int i = 0; i < NLD; ++i) {
        const size_t po = (size_t)i * DD * MMID;
        const size_t fo = (size_t)i * DD * FFH;
        // masked self-attention
        run_attention(stream, xdh, xdh,
                      dWq1_h + po, dbq1 + i * MMID, dWk1_h + po, dbk1 + i * MMID,
                      dWv1_h + po, dbv1 + i * MMID, dWo1_h + po, dbo1 + i * DD,
                      qb, kb, vb, ob, scores, probs, tmpD, /*causal=*/1);
        rmsnorm_kernel<<<BSR, 256, 0, stream>>>(xd, tmpD, dg1 + i * DD, dn1, dn1h);
        // cross-attention: q from dn1, k/v from enc_out
        run_attention(stream, dn1h, xeh,
                      dWq2_h + po, dbq2 + i * MMID, dWk2_h + po, dbk2 + i * MMID,
                      dWv2_h + po, dbv2 + i * MMID, dWo2_h + po, dbo2 + i * DD,
                      qb, kb, vb, ob, scores, probs, tmpD, /*causal=*/0);
        rmsnorm_kernel<<<BSR, 256, 0, stream>>>(dn1, tmpD, dg2 + i * DD, dn1, dn1h);
        run_ffn(stream, dn1h, dw1_h + fo, dw2_h + fo, dw3_h + fo, h1, h3, hm, tmpD);
        rmsnorm_kernel<<<BSR, 256, 0, stream>>>(dn1, tmpD, dg3 + i * DD, xd, xdh);
    }

    // ---- final vocab projection: [BSR,DD] x [DD,VV] + lin_b -> d_out (f32) ----
    launch_gemm(stream, xdh, linW_h, (float*)d_out, nullptr, linb,
                BSR, VV, DD, DD, DD, VV, VV, 0, 1.f, 1, 1, 0, 0, 0, 0, 0, 0);
}
